// sphericalunet_regression_confounded_75505525063934
// MI455X (gfx1250) — compile-verified, matrix operand order checked by reading, not run on hardware
//
#include <hip/hip_runtime.h>
#include <hip/hip_bf16.h>

typedef __attribute__((ext_vector_type(16))) __bf16 v16bf;
typedef __attribute__((ext_vector_type(8)))  float  v8f;

#define KNEI 19

__device__ __forceinline__ void cvt4(v16bf& f, int o, float4 v) {
    f[o + 0] = (__bf16)v.x;
    f[o + 1] = (__bf16)v.y;
    f[o + 2] = (__bf16)v.z;
    f[o + 3] = (__bf16)v.w;
}

// f32 -> bf16 weight repack (weights are invariant across all M tiles; pack
// once per conv so the GEMM's B fragments are raw 32-byte loads, no cvt).
__global__ void repack_bf16_kernel(const float* __restrict__ W,
                                   __bf16* __restrict__ Wbf, int n)
{
    const int i = blockIdx.x * blockDim.x + threadIdx.x;
    if (i < n) Wbf[i] = (__bf16)W[i];
}

// Main conv GEMM for Cin in {32,64,128,256} (KC = 19*Cin, multiple of 32).
// Activations in [V, B=4, Cin] layout; weights pre-packed bf16 [Cout, KC].
// One wave computes a 32 x (16*NT) output tile (MT=2 M-tiles x NT N-tiles):
// per K-step: 2 A fragments (4x b128 + pack each), NT B fragments (2x b128,
// no cvt), 2*NT back-to-back WMMAs.
//   A frag (16x32 bf16): lane l holds M=l%16, K=(l/16)*16+j.
//   B frag (32x16 bf16): lane l holds N=l%16, K=(l/16)*16+j.
//   C/D   (16x16 f32) : lane l, reg r -> M=(l/16)*8+r, N=l%16.
template <int NT>
__global__ void conv_main_kernel(const float*  __restrict__ x,    // [V,4,Cin]
                                 const int*    __restrict__ nbr,  // [V*19]
                                 const __bf16* __restrict__ Wbf,  // [Cout, KC]
                                 const float*  __restrict__ bias, // [Cout]
                                 float*        __restrict__ out,  // [V,4,Cout]
                                 int V, int Cin, int lgCin, int Cout, int KC)
{
    const int lane  = threadIdx.x & 31;
    const int mTile = blockIdx.x;          // 32 M-rows per block
    const int nBase = blockIdx.y * NT;
    const int M     = V * 4;
    const int half  = lane >> 4;
    const int l16   = lane & 15;

    const int* nrow[2];
    long xoff[2];
#pragma unroll
    for (int mt = 0; mt < 2; ++mt) {
        const int mA = mTile * 32 + mt * 16 + l16;
        const int vA = min(mA >> 2, V - 1);   // clamp: invalid rows never stored
        const int bA = mA & 3;
        nrow[mt] = nbr + vA * KNEI;
        xoff[mt] = ((long)vA * 4 + bA) * Cin;
        (void)xoff;
    }

    v8f zero = {};
    v8f acc[2][NT];
#pragma unroll
    for (int mt = 0; mt < 2; ++mt)
#pragma unroll
        for (int t = 0; t < NT; ++t) acc[mt][t] = zero;

    for (int kk = 0; kk < KC; kk += 32) {
        const int base = kk + half * 16;
        const int k    = base >> lgCin;       // neighbor slot (0..18)
        const int c0   = base & (Cin - 1);    // starting channel, mult of 16

        v16bf afrag[2];
#pragma unroll
        for (int mt = 0; mt < 2; ++mt) {
            const int mA = mTile * 32 + mt * 16 + l16;
            const int vA = min(mA >> 2, V - 1);
            const int bA = mA & 3;
            const int vn = nrow[mt][k];
            const float4* ap =
                (const float4*)(x + ((long)vn * 4 + bA) * Cin + c0);
            (void)vA;
            cvt4(afrag[mt], 0,  ap[0]);
            cvt4(afrag[mt], 4,  ap[1]);
            cvt4(afrag[mt], 8,  ap[2]);
            cvt4(afrag[mt], 12, ap[3]);
        }

#pragma unroll
        for (int t = 0; t < NT; ++t) {
            const int oB = (nBase + t) * 16 + l16;
            const v16bf bfrag = *(const v16bf*)(Wbf + (long)oB * KC + base);
#pragma unroll
            for (int mt = 0; mt < 2; ++mt) {
                acc[mt][t] = __builtin_amdgcn_wmma_f32_16x16x32_bf16(
                    false, afrag[mt], false, bfrag, (short)0, acc[mt][t],
                    false, false);
            }
        }
    }

#pragma unroll
    for (int mt = 0; mt < 2; ++mt)
#pragma unroll
        for (int t = 0; t < NT; ++t) {
            const int oB = (nBase + t) * 16 + l16;
            const float bb = bias[oB];
#pragma unroll
            for (int r = 0; r < 8; ++r) {
                const int m = mTile * 32 + mt * 16 + half * 8 + r;
                if (m < M) out[(long)m * Cout + oB] = acc[mt][t][r] + bb;
            }
        }
}

// First conv: Cin=4, KC=76 (not a multiple of 32, ~2% of FLOPs). Input keeps
// the reference [V, Cin=4, B=4] layout; output written [V, B, 32].
__global__ void conv_first_kernel(const float* __restrict__ x,    // [V,4,4]
                                  const int*   __restrict__ nbr,
                                  const float* __restrict__ W,    // [32,76]
                                  const float* __restrict__ bias,
                                  float*       __restrict__ out,  // [V,4,32]
                                  int V)
{
    const int KC = 76, Cout = 32;
    const int lane  = threadIdx.x & 31;
    const int mTile = blockIdx.x;
    const int M     = V * 4;
    const int half  = lane >> 4;
    const int l16   = lane & 15;

    const int mA = mTile * 16 + l16;
    const int vA = min(mA >> 2, V - 1);
    const int bA = mA & 3;
    const int* nrow = nbr + vA * KNEI;

    v8f zero = {};
    v8f acc[2] = {zero, zero};

    for (int kk = 0; kk < KC; kk += 32) {
        v16bf afrag;
#pragma unroll
        for (int j = 0; j < 16; ++j) {
            const int i = kk + half * 16 + j;
            float av = 0.0f;
            if (i < KC) {
                const int k = i >> 2, c = i & 3;
                const int vn = nrow[k];
                av = x[((long)vn * 4 + c) * 4 + bA];
            }
            afrag[j] = (__bf16)av;
        }
#pragma unroll
        for (int t = 0; t < 2; ++t) {
            const int oB = t * 16 + l16;
            v16bf bfrag;
#pragma unroll
            for (int j = 0; j < 16; ++j) {
                const int i = kk + half * 16 + j;
                bfrag[j] = (__bf16)((i < KC) ? W[(long)oB * KC + i] : 0.0f);
            }
            acc[t] = __builtin_amdgcn_wmma_f32_16x16x32_bf16(
                false, afrag, false, bfrag, (short)0, acc[t], false, false);
        }
    }

#pragma unroll
    for (int t = 0; t < 2; ++t) {
        const int oB = t * 16 + l16;
        const float bb = bias[oB];
#pragma unroll
        for (int r = 0; r < 8; ++r) {
            const int m = mTile * 16 + half * 8 + r;
            if (m < M) out[(long)m * Cout + oB] = acc[t][r] + bb;
        }
    }
}

// Per-channel mean and rsqrt(var+eps) over (V,B). Layout [V*4, C].
__global__ void bn_stats_kernel(const float* __restrict__ x, int V, int C,
                                float* __restrict__ stats)
{
    __shared__ float s_sum[256];
    __shared__ float s_sq[256];
    const int c = blockIdx.x;
    const int N = V * 4;
    float s = 0.f, s2 = 0.f;
    for (int i = threadIdx.x; i < N; i += blockDim.x) {
        const float val = x[(long)i * C + c];
        s += val; s2 += val * val;
    }
    s_sum[threadIdx.x] = s;
    s_sq[threadIdx.x]  = s2;
    __syncthreads();
    for (int off = 128; off > 0; off >>= 1) {
        if ((int)threadIdx.x < off) {
            s_sum[threadIdx.x] += s_sum[threadIdx.x + off];
            s_sq[threadIdx.x]  += s_sq[threadIdx.x + off];
        }
        __syncthreads();
    }
    if (threadIdx.x == 0) {
        const float inv = 1.0f / (float)N;
        const float mu  = s_sum[0] * inv;
        const float var = s_sq[0] * inv - mu * mu;
        stats[c]     = mu;
        stats[C + c] = rsqrtf(var + 1e-5f);
    }
}

__global__ void bn_lrelu_kernel(float* __restrict__ x,
                                const float* __restrict__ stats,
                                const float* __restrict__ gamma,
                                const float* __restrict__ beta,
                                int V, int C)
{
    const long total = (long)V * C * 4;
    const long idx = (long)blockIdx.x * blockDim.x + threadIdx.x;
    if (idx >= total) return;
    const int c = (int)(idx % C);
    const float y = gamma[c] * (x[idx] - stats[c]) * stats[C + c] + beta[c];
    x[idx] = (y > 0.f) ? y : 0.2f * y;
}

// Layout [V,B,C]: out[(vp*4+b)*C + c] = mean_k x[(nbr[vp*19+k]*4+b)*C + c]
__global__ void pool_kernel(const float* __restrict__ x,
                            const int*   __restrict__ nbr,
                            float*       __restrict__ out,
                            int Vp, int C)
{
    const long total = (long)Vp * C * 4;
    const long idx = (long)blockIdx.x * blockDim.x + threadIdx.x;
    if (idx >= total) return;
    const int c  = (int)(idx % C);
    const int r  = (int)(idx / C);
    const int b  = r & 3;
    const int vp = r >> 2;
    float s = 0.f;
    for (int k = 0; k < KNEI; ++k) {
        const int vn = nbr[vp * KNEI + k];
        s += x[((long)vn * 4 + b) * C + c];
    }
    out[idx] = s * (1.0f / 19.0f);
}

// x in [642, B, 256] layout.
// out[b] = sum_{v,c} x[(v*4+b)*256+c]*wfc[v*256+c]
//        + sum_j relu(m[b]*wm[j]+bm[j])*wfc[642*256+j] + bfc
__global__ void final_fc_kernel(const float* __restrict__ x,
                                const float* __restrict__ m,
                                const float* __restrict__ wm,
                                const float* __restrict__ bm,
                                const float* __restrict__ wfc,
                                const float* __restrict__ bfc,
                                float*       __restrict__ out)
{
    __shared__ float s_sum[256];
    const int b = blockIdx.x;
    const int total = 642 * 256;
    float s = 0.f;
    for (int i = threadIdx.x; i < total; i += blockDim.x) {
        const int v = i >> 8;
        const int c = i & 255;
        s += x[((long)v * 4 + b) * 256 + c] * wfc[i];
    }
    s_sum[threadIdx.x] = s;
    __syncthreads();
    for (int off = 128; off > 0; off >>= 1) {
        if ((int)threadIdx.x < off) s_sum[threadIdx.x] += s_sum[threadIdx.x + off];
        __syncthreads();
    }
    if (threadIdx.x == 0) {
        float t = s_sum[0];
        const float mb = m[b];
        for (int j = 0; j < 4; ++j) {
            float mm = mb * wm[j] + bm[j];
            mm = mm > 0.f ? mm : 0.f;
            t += mm * wfc[total + j];
        }
        out[b] = t + bfc[0];
    }
}

static void run_bn(float* buf, const float* gamma, const float* beta,
                   float* stats, int V, int C, hipStream_t stream)
{
    bn_stats_kernel<<<C, 256, 0, stream>>>(buf, V, C, stats);
    const long total = (long)V * C * 4;
    bn_lrelu_kernel<<<(int)((total + 255) / 256), 256, 0, stream>>>(
        buf, stats, gamma, beta, V, C);
}

static void run_conv_main(const float* in, const int* nbr,
                          const float* W, const float* bias,
                          __bf16* wscratch, float* outbuf,
                          int V, int Cin, int lgCin, int Cout,
                          hipStream_t stream)
{
    const int KC = KNEI * Cin;
    const int nW = Cout * KC;
    repack_bf16_kernel<<<(nW + 255) / 256, 256, 0, stream>>>(W, wscratch, nW);

    const int M = V * 4;
    const int mtiles = (M + 31) / 32;
    if (Cout >= 64) {
        dim3 grid(mtiles, Cout / 64);
        conv_main_kernel<4><<<grid, 32, 0, stream>>>(in, nbr, wscratch, bias,
                                                     outbuf, V, Cin, lgCin,
                                                     Cout, KC);
    } else {
        dim3 grid(mtiles, Cout / 32);
        conv_main_kernel<2><<<grid, 32, 0, stream>>>(in, nbr, wscratch, bias,
                                                     outbuf, V, Cin, lgCin,
                                                     Cout, KC);
    }
}

extern "C" void kernel_launch(void* const* d_in, const int* in_sizes, int n_in,
                              void* d_out, int out_size, void* d_ws, size_t ws_size,
                              hipStream_t stream)
{
    (void)in_sizes; (void)n_in; (void)out_size; (void)ws_size;

    const float* x  = (const float*)d_in[0];
    const float* m  = (const float*)d_in[1];
    const int*   n0 = (const int*)d_in[2];
    const int*   n1 = (const int*)d_in[3];
    const int*   n2 = (const int*)d_in[4];
    const int*   n3 = (const int*)d_in[5];

    const float* wm  = (const float*)d_in[38];
    const float* bm  = (const float*)d_in[39];
    const float* wfc = (const float*)d_in[40];
    const float* bfc = (const float*)d_in[41];

    // Workspace: two ping-pong activation buffers [V,B,C] + BN stats + bf16
    // weight scratch. Max activation: 40962*4*32 = 5,243,136 floats; max
    // weight: 256*4864 = 1,245,184 bf16.
    const long BUF = 5250048;
    float*  bufA  = (float*)d_ws;
    float*  bufB  = bufA + BUF;
    float*  stats = bufB + BUF;            // 2*256 floats
    __bf16* wbf   = (__bf16*)(stats + 1024);

    long t;

    // ---- Level 0: V=40962, 4 -> 32 -> 32 (neighbors n0) ----
    {
        const int M = 40962 * 4;
        conv_first_kernel<<<(M + 15) / 16, 32, 0, stream>>>(
            x, n0, (const float*)d_in[6], (const float*)d_in[7], bufA, 40962);
        run_bn(bufA, (const float*)d_in[8], (const float*)d_in[9], stats,
               40962, 32, stream);
    }
    run_conv_main(bufA, n0, (const float*)d_in[10], (const float*)d_in[11],
                  wbf, bufB, 40962, 32, 5, 32, stream);
    run_bn(bufB, (const float*)d_in[12], (const float*)d_in[13], stats, 40962, 32, stream);
    t = (long)10242 * 32 * 4;
    pool_kernel<<<(int)((t + 255) / 256), 256, 0, stream>>>(bufB, n0, bufA, 10242, 32);

    // ---- Level 1: V=10242, 32 -> 64 -> 64 (neighbors n1) ----
    run_conv_main(bufA, n1, (const float*)d_in[14], (const float*)d_in[15],
                  wbf, bufB, 10242, 32, 5, 64, stream);
    run_bn(bufB, (const float*)d_in[16], (const float*)d_in[17], stats, 10242, 64, stream);
    run_conv_main(bufB, n1, (const float*)d_in[18], (const float*)d_in[19],
                  wbf, bufA, 10242, 64, 6, 64, stream);
    run_bn(bufA, (const float*)d_in[20], (const float*)d_in[21], stats, 10242, 64, stream);
    t = (long)2562 * 64 * 4;
    pool_kernel<<<(int)((t + 255) / 256), 256, 0, stream>>>(bufA, n1, bufB, 2562, 64);

    // ---- Level 2: V=2562, 64 -> 128 -> 128 (neighbors n2) ----
    run_conv_main(bufB, n2, (const float*)d_in[22], (const float*)d_in[23],
                  wbf, bufA, 2562, 64, 6, 128, stream);
    run_bn(bufA, (const float*)d_in[24], (const float*)d_in[25], stats, 2562, 128, stream);
    run_conv_main(bufA, n2, (const float*)d_in[26], (const float*)d_in[27],
                  wbf, bufB, 2562, 128, 7, 128, stream);
    run_bn(bufB, (const float*)d_in[28], (const float*)d_in[29], stats, 2562, 128, stream);
    t = (long)642 * 128 * 4;
    pool_kernel<<<(int)((t + 255) / 256), 256, 0, stream>>>(bufB, n2, bufA, 642, 128);

    // ---- Level 3: V=642, 128 -> 256 -> 256 (neighbors n3) ----
    run_conv_main(bufA, n3, (const float*)d_in[30], (const float*)d_in[31],
                  wbf, bufB, 642, 128, 7, 256, stream);
    run_bn(bufB, (const float*)d_in[32], (const float*)d_in[33], stats, 642, 256, stream);
    run_conv_main(bufB, n3, (const float*)d_in[34], (const float*)d_in[35],
                  wbf, bufA, 642, 256, 8, 256, stream);
    run_bn(bufA, (const float*)d_in[36], (const float*)d_in[37], stats, 642, 256, stream);

    // ---- Final FC over [642,B,256] + m-branch ----
    final_fc_kernel<<<4, 256, 0, stream>>>(bufA, m, wm, bm, wfc, bfc, (float*)d_out);
}
